// TopicFormer_77154792506109
// MI455X (gfx1250) — matrix operand order repaired and checked by reading, not compiled, not run on hardware
//
#include <hip/hip_runtime.h>
#include <hip/hip_bf16.h>
#include <cstdint>
#include <cstddef>

// ---------------- problem constants (from reference setup_inputs) ----------
#define NB   4      // batch N
#define LL   4096   // L (= S)
#define CC   256    // C
#define KK   100    // topics / seeds
#define DTT  96     // topic feature dim
#define NH   8      // heads
#define HD   32     // head dim
#define NSAMP 6
#define NTT  2

typedef __bf16 bf16;
typedef bf16 v16bf __attribute__((ext_vector_type(16)));
typedef bf16 v8bf  __attribute__((ext_vector_type(8)));
typedef float v8f  __attribute__((ext_vector_type(8)));

// ---- CDNA5 async global->LDS copy (ASYNCcnt) if the toolchain exposes it --
#if defined(__has_builtin)
# if __has_builtin(__builtin_amdgcn_global_load_async_to_lds_b128) && \
     __has_builtin(__builtin_amdgcn_s_wait_asynccnt)
#  define HAVE_ASYNC_LDS 1
# endif
#endif
#ifndef HAVE_ASYNC_LDS
# define HAVE_ASYNC_LDS 0
#endif

#if HAVE_ASYNC_LDS
// builtin signature (from clang diagnostic): (int4 AS1*, int4 AS3*, imm, imm)
typedef int v4i __attribute__((vector_size(16)));
typedef __attribute__((address_space(1))) v4i gv4i;
typedef __attribute__((address_space(3))) v4i lv4i;
static __device__ inline void async_cp16(const bf16* g, bf16* l) {
  __builtin_amdgcn_global_load_async_to_lds_b128((gv4i*)(void*)g, (lv4i*)(void*)l, 0, 0);
}
#endif

// =================== WMMA bf16 GEMM: C[M,N] (+)= alpha * A @ B ============
// A: bf16 M x K row-major (lda).  BT: bf16 N x K row-major (ldbt) == B^T.
// C: f32. 128x128 block tile, BK=64, 8 wave32 waves each 64x32 (8 wmma/kk).
// Interior tiles: double-buffered async global->LDS pipeline (ASYNCcnt).
#define BM 128
#define BN 128
#define BK 64
#define BKP 72   // padded LDS row stride (bf16 elems) -> conflict-free frags

__global__ void __launch_bounds__(256) gemm_kernel(
    const bf16* __restrict__ A, long lda, long sA,
    const bf16* __restrict__ BT, long ldbt, long sBT,
    float* __restrict__ C, long ldc, long sC,
    int M, int N, int Kd, float alpha, int accum)
{
#if __has_builtin(__builtin_amdgcn_s_wait_tensorcnt)
  __builtin_amdgcn_s_wait_tensorcnt(0);
#endif
  __shared__ __attribute__((aligned(16))) bf16 As[2][BM * BKP];
  __shared__ __attribute__((aligned(16))) bf16 Bs[2][BN * BKP];
  const int tid  = threadIdx.x;
  const int wave = tid >> 5, lane = tid & 31;
  const int waveM = wave >> 2, waveN = wave & 3;   // 2 x 4 wave grid
  const int hsel = lane >> 4, r = lane & 15;
  const int tileM = blockIdx.y * BM, tileN = blockIdx.x * BN;
  const bf16* Ap = A + (long)blockIdx.z * sA;
  const bf16* Bp = BT + (long)blockIdx.z * sBT;
  float* Cp = C + (long)blockIdx.z * sC;
  v8f acc[4][2] = {};
  const int nk = (Kd + BK - 1) / BK;

  // ---- one K-step of WMMA work from LDS buffer `buf` ----
  auto compute = [&](int buf) {
#pragma unroll
    for (int kk = 0; kk < 2; ++kk) {
      const int kf = kk * 32;
      v16bf afr[4], bfr[2];
      // A fragment (16x32 bf16): lane<16 -> K {0..7,16..23}; lane>=16 -> +8
#pragma unroll
      for (int wm = 0; wm < 4; ++wm) {
        int rowB = (waveM * 64 + wm * 16 + r) * BKP;
        int run0 = kf + hsel * 8;
        union { v16bf v; v8bf h[2]; } u;
        u.h[0] = *(const v8bf*)&As[buf][rowB + run0];
        u.h[1] = *(const v8bf*)&As[buf][rowB + run0 + 16];
        afr[wm] = u.v;
      }
      // B fragment (32x16 bf16): col per lane, lane<16 -> K {0..7,16..23}; lane>=16 -> +8
#pragma unroll
      for (int wn = 0; wn < 2; ++wn) {
        int colB = (waveN * 32 + wn * 16 + r) * BKP;
        int ks = kf + hsel * 8;
        union { v16bf v; v8bf h[2]; } u;
        u.h[0] = *(const v8bf*)&Bs[buf][colB + ks];
        u.h[1] = *(const v8bf*)&Bs[buf][colB + ks + 16];
        bfr[wn] = u.v;
      }
#pragma unroll
      for (int wm = 0; wm < 4; ++wm)
#pragma unroll
        for (int wn = 0; wn < 2; ++wn)
          acc[wm][wn] = __builtin_amdgcn_wmma_f32_16x16x32_bf16(
              false, afr[wm], false, bfr[wn], (short)0, acc[wm][wn], false, false);
    }
  };

#if HAVE_ASYNC_LDS
  const bool pipeAsync = (tileM + BM <= M) && (tileN + BN <= N) && (Kd % BK == 0);
  if (pipeAsync) {
    // ---- double-buffered async pipeline: copy tile k+1 during tile k ----
    auto stage = [&](int kt, int buf) {
      const int k0 = kt * BK;
#pragma unroll
      for (int i = 0; i < 4; ++i) {
        int q = tid + i * 256; int row = q >> 3; int c8 = (q & 7) << 3;
        async_cp16(Ap + (long)(tileM + row) * lda + (k0 + c8), &As[buf][row * BKP + c8]);
        async_cp16(Bp + (long)(tileN + row) * ldbt + (k0 + c8), &Bs[buf][row * BKP + c8]);
      }
    };
    stage(0, 0);
    for (int kt = 0; kt < nk; ++kt) {
      const int cur = kt & 1;
      const bool more = (kt + 1 < nk);
      if (more) {
        stage(kt + 1, cur ^ 1);
        __builtin_amdgcn_s_wait_asynccnt(8);   // tile k landed; k+1 in flight
      } else {
        __builtin_amdgcn_s_wait_asynccnt(0);
      }
      __syncthreads();
      compute(cur);
      __syncthreads();
    }
  } else
#endif
  {
    // ---- manual staging with bounds (edge tiles / fallback) ----
    for (int kt = 0; kt < nk; ++kt) {
      const int k0 = kt * BK;
      if (kt + 1 < nk) {
        int pr = tileM + (tid & 127);
        if (pr < M) __builtin_prefetch(Ap + (long)pr * lda + k0 + BK, 0, 1);
        int pn = tileN + (tid & 127);
        if (pn < N) __builtin_prefetch(Bp + (long)pn * ldbt + k0 + BK, 0, 1);
      }
#pragma unroll
      for (int i = 0; i < 4; ++i) {
        int q = tid + i * 256; int row = q >> 3; int c8 = (q & 7) << 3;
        int gk = k0 + c8;
        v8bf va = {};
        int gr = tileM + row;
        if (gr < M) {
          if (gk + 7 < Kd) va = *(const v8bf*)(Ap + (long)gr * lda + gk);
          else {
            const bf16* prow = Ap + (long)gr * lda;
            for (int e = 0; e < 8; ++e) if (gk + e < Kd) va[e] = prow[gk + e];
          }
        }
        *(v8bf*)&As[0][row * BKP + c8] = va;
        v8bf vb = {};
        int gn = tileN + row;
        if (gn < N) {
          if (gk + 7 < Kd) vb = *(const v8bf*)(Bp + (long)gn * ldbt + gk);
          else {
            const bf16* prow = Bp + (long)gn * ldbt;
            for (int e = 0; e < 8; ++e) if (gk + e < Kd) vb[e] = prow[gk + e];
          }
        }
        *(v8bf*)&Bs[0][row * BKP + c8] = vb;
      }
      __syncthreads();
      compute(0);
      __syncthreads();
    }
  }

  // epilogue: C/D layout -> lane r = col, VGPR e = row (e + 8*hsel)
#pragma unroll
  for (int wm = 0; wm < 4; ++wm) {
#pragma unroll
    for (int wn = 0; wn < 2; ++wn) {
      int gn = tileN + waveN * 32 + wn * 16 + r;
      if (gn >= N) continue;
      int mb = tileM + waveM * 64 + wm * 16 + hsel * 8;
#pragma unroll
      for (int e = 0; e < 8; ++e) {
        int gm = mb + e;
        if (gm < M) {
          float val = alpha * acc[wm][wn][e];
          long off = (long)gm * ldc + gn;
          if (accum) val += Cp[off];
          Cp[off] = val;
        }
      }
    }
  }
}

// =================== dtype conversion kernels ==============================
__global__ void cvt_bf16_kernel(const float* __restrict__ in, bf16* __restrict__ out, long total) {
  long i = (long)blockIdx.x * blockDim.x + threadIdx.x;
  if (i < total) out[i] = (bf16)in[i];
}

// transpose + convert: in (nb, R, Cc) f32 -> out (nb, Cc, R) bf16
__global__ void tcvt_bf16_kernel(const float* __restrict__ in, bf16* __restrict__ out,
                                 int R, int Cc, int nb) {
  long i = (long)blockIdx.x * blockDim.x + threadIdx.x;
  long total = (long)nb * R * Cc;
  if (i >= total) return;
  int c = i % Cc; long t = i / Cc; int rr = t % R; int n = t / R;
  out[((long)n * Cc + c) * R + rr] = (bf16)in[i];
}

// =================== elementwise / attention helper kernels ================
__global__ void elu1_mask_kernel(float* __restrict__ x, const float* __restrict__ mask,
                                 long rows, int C) {
  long i = (long)blockIdx.x * blockDim.x + threadIdx.x;
  if (i >= rows * (long)C) return;
  float v = x[i];
  float q = v > 0.f ? v + 1.f : expf(v);   // elu(v)+1
  if (mask) q *= mask[i / C];
  x[i] = q;
}

__global__ void mask_mul_kernel(float* __restrict__ x, const float* __restrict__ mask,
                                long rows, int C) {
  long i = (long)blockIdx.x * blockDim.x + threadIdx.x;
  if (i >= rows * (long)C) return;
  x[i] *= mask[i / C];
}

__global__ void add_inplace_kernel(float* __restrict__ x, const float* __restrict__ y, long total) {
  long i = (long)blockIdx.x * blockDim.x + threadIdx.x;
  if (i < total) x[i] += y[i];
}

// KV[n,h,di,dj] = sum_s K[n,s,h,di]*V[n,s,h,dj];  Ksum[n,h,d] = sum_s K
__global__ void __launch_bounds__(256) kv_ksum_kernel(
    const float* __restrict__ Kb, const float* __restrict__ Vb,
    float* __restrict__ KV, float* __restrict__ Ksum, int slen)
{
  __shared__ float Kt[32][33];
  __shared__ float Vt[32][33];
  const int n = blockIdx.y, h = blockIdx.x, tid = threadIdx.x;
  const float* Kp = Kb + (long)n * slen * CC + h * HD;
  const float* Vp = Vb + (long)n * slen * CC + h * HD;
  const int p0 = tid * 4;
  const int di = p0 >> 5, dj0 = p0 & 31;
  float acc[4] = {0.f, 0.f, 0.f, 0.f};
  float ks = 0.f;
  const int nt = (slen + 31) / 32;
  for (int st = 0; st < nt; ++st) {
#pragma unroll
    for (int j = 0; j < 4; ++j) {
      int q = tid * 4 + j; int srow = q >> 5, scol = q & 31;
      int s = st * 32 + srow;
      float kv = 0.f, vv = 0.f;
      if (s < slen) { kv = Kp[(long)s * CC + scol]; vv = Vp[(long)s * CC + scol]; }
      Kt[srow][scol] = kv; Vt[srow][scol] = vv;
    }
    __syncthreads();
    for (int ss = 0; ss < 32; ++ss) {
      float kd = Kt[ss][di];
#pragma unroll
      for (int j = 0; j < 4; ++j) acc[j] += kd * Vt[ss][dj0 + j];
      if (tid < 32) ks += Kt[ss][tid];
    }
    __syncthreads();
  }
  float* outp = KV + ((long)(n * NH + h) * HD + di) * HD + dj0;
#pragma unroll
  for (int j = 0; j < 4; ++j) outp[j] = acc[j];
  if (tid < 32) Ksum[(long)(n * NH + h) * HD + tid] = ks;
}

// msg[n,l,h,j] = (sum_d Q*KV) / (sum_d Q*Ksum + eps)   (s-scalings cancel)
__global__ void __launch_bounds__(256) msg_kernel(
    const float* __restrict__ Q, const float* __restrict__ KV,
    const float* __restrict__ Ks, float* __restrict__ msg,
    bf16* __restrict__ msgb, int llen)
{
  __shared__ float KVs[NH * HD * HD];
  __shared__ float Kss[CC];
  __shared__ float Qs[CC];
  const int n = blockIdx.y, t = threadIdx.x;
  for (int i = t; i < NH * HD * HD; i += 256) KVs[i] = KV[(long)n * NH * HD * HD + i];
  Kss[t] = Ks[(long)n * CC + t];
  __syncthreads();
  const int h = t >> 5, j = t & 31;
  for (int rr = 0; rr < 16; ++rr) {
    int row = blockIdx.x * 16 + rr;
    float qv = 0.f;
    if (row < llen) qv = Q[((long)n * llen + row) * CC + t];
    Qs[t] = qv;
    __syncthreads();
    if (row < llen) {
      float zden = 0.f, m = 0.f;
      const float* kvh = &KVs[h * HD * HD];
#pragma unroll 4
      for (int d = 0; d < HD; ++d) {
        float qd = Qs[h * HD + d];
        zden += qd * Kss[h * HD + d];
        m += qd * kvh[d * HD + j];
      }
      float out = m / (zden + 1e-6f);
      long off = ((long)n * llen + row) * CC + h * HD + j;
      msg[off] = out;
      msgb[off] = (bf16)out;   // fused cvt for the Wm GEMM
    }
    __syncthreads();
  }
}

// layernorm; optionally emits bf16 copy for a following GEMM
__global__ void __launch_bounds__(256) layernorm_kernel(float* __restrict__ x,
                                                        bf16* __restrict__ xb, int W) {
  __shared__ float rs[256], rs2[256];
  float* p = x + (long)blockIdx.x * W;
  const int t = threadIdx.x;
  float s = 0.f, s2 = 0.f;
  for (int i = t; i < W; i += 256) { float v = p[i]; s += v; s2 += v * v; }
  rs[t] = s; rs2[t] = s2; __syncthreads();
  for (int o = 128; o > 0; o >>= 1) {
    if (t < o) { rs[t] += rs[t + o]; rs2[t] += rs2[t + o]; }
    __syncthreads();
  }
  float mean = rs[0] / W;
  float var  = rs2[0] / W - mean * mean;
  float inv  = rsqrtf(var + 1e-5f);
  for (int i = t; i < W; i += 256) {
    float v = (p[i] - mean) * inv;
    p[i] = v;
    if (xb) xb[(long)blockIdx.x * W + i] = (bf16)v;
  }
}

// relu + bf16 copy (feeds the W2 GEMM)
__global__ void relu_cvt_kernel(float* __restrict__ x, bf16* __restrict__ xb, long total) {
  long i = (long)blockIdx.x * blockDim.x + threadIdx.x;
  if (i < total) { float v = fmaxf(x[i], 0.f); x[i] = v; xb[i] = (bf16)v; }
}

// =================== topic / pooling / misc kernels ========================
__global__ void transpose_kernel(const float* __restrict__ in, float* __restrict__ out,
                                 int R, int Cc, int nb) {
  long i = (long)blockIdx.x * blockDim.x + threadIdx.x;
  long total = (long)nb * R * Cc;
  if (i >= total) return;
  int c = i % Cc; long t = i / Cc; int rr = t % R; int n = t / R;
  out[((long)n * Cc + c) * R + rr] = in[((long)n * R + rr) * Cc + c];
}

__global__ void floor_scale_kernel(float* __restrict__ x, long total) {
  long i = (long)blockIdx.x * blockDim.x + threadIdx.x;
  if (i < total) x[i] = floorf(x[i] * (1.f / 16.f));   // sqrt(C)=16
}

__global__ void __launch_bounds__(256) tm_soft_kernel(float* __restrict__ tm, int* __restrict__ tmidx) {
  __shared__ float S[KK * KK];
  __shared__ float cmax[KK], csum[KK], rmax[KK], rsum[KK];
  const int n = blockIdx.x, t = threadIdx.x, tot = KK * KK;
  float* g = tm + (long)n * tot;
  for (int i = t; i < tot; i += 256) S[i] = g[i];
  __syncthreads();
  if (t < KK) {                       // softmax over axis=1 (rows m) per col l
    float mx = -1e30f; for (int m = 0; m < KK; ++m) mx = fmaxf(mx, S[m * KK + t]);
    float sm = 0.f;    for (int m = 0; m < KK; ++m) sm += expf(S[m * KK + t] - mx);
    cmax[t] = mx; csum[t] = sm;
  } else if (t >= 128 && t < 128 + KK) {   // softmax over axis=2 per row m
    int m = t - 128;
    float mx = -1e30f; for (int l = 0; l < KK; ++l) mx = fmaxf(mx, S[m * KK + l]);
    float sm = 0.f;    for (int l = 0; l < KK; ++l) sm += expf(S[m * KK + l] - mx);
    rmax[m] = mx; rsum[m] = sm;
  }
  __syncthreads();
  for (int i = t; i < tot; i += 256) {
    int m = i / KK, l = i % KK;
    float v = S[i];
    float p = (expf(v - cmax[l]) / csum[l]) * (expf(v - rmax[m]) / rsum[m]);
    S[i] = p; g[i] = p;
  }
  __syncthreads();
  if (t < KK) {
    float best = -1e30f; int bi = 0;
    for (int l = 0; l < KK; ++l) { float v = S[t * KK + l]; if (v > best) { best = v; bi = l; } }
    tmidx[n * KK + t] = bi;
  }
}

__global__ void gather_t1_kernel(const float* __restrict__ t1s, const int* __restrict__ tmidx,
                                 float* __restrict__ t1g) {
  long i = (long)blockIdx.x * blockDim.x + threadIdx.x;
  if (i >= (long)NB * KK * DTT) return;
  int d = i % DTT; long t = i / DTT; int m = t % KK; int n = t / KK;
  int idx = tmidx[(NB - 1) * KK + m];   // reference quirk: tm_idx[-1] for all n
  t1g[i] = t1s[((long)n * KK + idx) * DTT + d];
}

__global__ void topic_pool_kernel(const float* __restrict__ t0s, const float* __restrict__ t1g,
                                  float* __restrict__ tp) {
  long i = (long)blockIdx.x * blockDim.x + threadIdx.x;
  if (i >= (long)NB * KK * DTT) return;
  int o = i % DTT; long t = i / DTT;
  int j0 = 2 * o, j1 = 2 * o + 1;
  float a = (j0 < DTT) ? t0s[t * DTT + j0] : t1g[t * DTT + j0 - DTT];
  float b = (j1 < DTT) ? t0s[t * DTT + j1] : t1g[t * DTT + j1 - DTT];
  tp[i] = 0.5f * (a + b);
}

__global__ void pool_kernel(const float* __restrict__ in, float* __restrict__ out,
                            long rows, int nin, int nout) {
  long i = (long)blockIdx.x * blockDim.x + threadIdx.x;
  if (i >= rows * nout) return;
  int o = i % nout; long r = i / nout;
  int s = (int)(((long)o * nin) / nout);
  int e = (int)((((long)(o + 1)) * nin + nout - 1) / nout);
  const float* p = in + r * nin;
  float acc = 0.f;
  for (int k = s; k < e; ++k) acc += p[k];
  out[i] = acc / (float)(e - s);
}

__global__ void concat_seeds_kernel(const float* __restrict__ sp, const float* __restrict__ tp,
                                    float* __restrict__ sf) {
  long i = (long)blockIdx.x * blockDim.x + threadIdx.x;
  if (i >= (long)NB * KK * CC) return;
  int c = i % CC; long t = i / CC;
  sf[i] = (c < 160) ? sp[t * 160 + c] : tp[t * DTT + (c - 160)];
}

__global__ void bcast_seeds_kernel(const float* __restrict__ st, float* __restrict__ seeds) {
  long i = (long)blockIdx.x * blockDim.x + threadIdx.x;
  if (i >= (long)NB * KK * CC) return;
  seeds[i] = st[i % ((long)KK * CC)];
}

// =================== dmatrix -> prob / theta / topk / refinement ===========
__global__ void prob_theta_kernel(const float* __restrict__ dm, int* __restrict__ fidx,
                                  float* __restrict__ th0, float* __restrict__ th1) {
  long r = (long)blockIdx.x * blockDim.x + threadIdx.x;
  if (r >= (long)NB * 2 * LL) return;
  int n = (int)(r / (2 * LL)); int l = (int)(r % (2 * LL));
  const float* p = dm + r * KK;
  float mx = p[0]; int bi = 0;
  for (int k = 1; k < KK; ++k) { float v = p[k]; if (v > mx) { mx = v; bi = k; } }
  float sm = 0.f;
  for (int k = 0; k < KK; ++k) sm += expf(p[k] - mx);
  float inv = 1.f / sm;
  float* th = (l < LL) ? th0 : th1;
  for (int k = 0; k < KK; ++k) atomicAdd(&th[n * KK + k], expf(p[k] - mx) * inv);
  fidx[r] = bi;
}

__global__ void theta_topk_kernel(const float* __restrict__ th0, const float* __restrict__ th1,
                                  int* __restrict__ inds) {
  if (threadIdx.x != 0) return;
  int n = blockIdx.x;
  float a[KK], b[KK], t[KK];
  float s0 = 0.f, s1 = 0.f;
  for (int k = 0; k < KK; ++k) { a[k] = th0[n * KK + k]; s0 += fabsf(a[k]); }
  for (int k = 0; k < KK; ++k) { b[k] = th1[n * KK + k]; s1 += fabsf(b[k]); }
  s0 = fmaxf(s0, 1e-12f); s1 = fmaxf(s1, 1e-12f);
  for (int k = 0; k < KK; ++k) t[k] = (a[k] / s0) * (b[k] / s1);
  bool used[KK];
  for (int k = 0; k < KK; ++k) used[k] = false;
  for (int kk = 0; kk < NSAMP; ++kk) {
    float best = -1e30f; int bi = 0;
    for (int k = 0; k < KK; ++k)
      if (!used[k] && t[k] > best) { best = t[k]; bi = k; }
    used[bi] = true;
    inds[n * NSAMP + kk] = bi;
  }
}

__global__ void partition_kernel(const int* __restrict__ fidx, int fofs,
                                 const int* __restrict__ inds, int kk,
                                 int* __restrict__ ord, float* __restrict__ nm,
                                 int* __restrict__ cnt) {
  if (threadIdx.x != 0) return;
  int n = blockIdx.x;
  int ind = inds[n * NSAMP + kk];
  const int* fp = fidx + (long)n * 2 * LL + fofs;
  int* op = ord + (long)n * LL;
  int c = 0;
  for (int l = 0; l < LL; ++l) if (fp[l] == ind) op[c++] = l;   // stable: selected first
  cnt[n] = c;
  int c2 = c;
  for (int l = 0; l < LL; ++l) if (fp[l] != ind) op[c2++] = l;
  for (int l = 0; l < LL; ++l) nm[(long)n * LL + l] = (l < c) ? 1.f : 0.f;
}

__global__ void cond_kernel(const int* __restrict__ c0, const int* __restrict__ c1,
                            float* __restrict__ cond) {
  int t0 = 0, t1 = 0;
  for (int n = 0; n < NB; ++n) { t0 += c0[n]; t1 += c1[n]; }
  cond[0] = (t0 > 0 && t1 > 0) ? 1.f : 0.f;
}

__global__ void ssum_kernel(float* __restrict__ ssum, const int* __restrict__ fidx, int fofs,
                            const int* __restrict__ inds, int kk) {
  long i = (long)blockIdx.x * blockDim.x + threadIdx.x;
  if (i >= (long)NB * LL) return;
  int n = (int)(i / LL); int l = (int)(i % LL);
  if (fidx[(long)n * 2 * LL + fofs + l] == inds[n * NSAMP + kk]) ssum[i] += 1.f;
}

__global__ void gather_kernel(float* __restrict__ nf, const float* __restrict__ feat,
                              const int* __restrict__ ord, const float* __restrict__ nm) {
  long i = (long)blockIdx.x * blockDim.x + threadIdx.x;
  if (i >= (long)NB * LL * CC) return;
  int c = i % CC; long t = i / CC; int li = (int)(t % LL); int n = (int)(t / LL);
  int src = ord[(long)n * LL + li];
  nf[i] = feat[((long)n * LL + src) * CC + c] * nm[(long)n * LL + li];
}

__global__ void scatter_kernel(float* __restrict__ upd, const float* __restrict__ nf,
                               const int* __restrict__ ord, const float* __restrict__ nm,
                               const float* __restrict__ cond) {
  long i = (long)blockIdx.x * blockDim.x + threadIdx.x;
  if (i >= (long)NB * LL * CC) return;
  int c = i % CC; long t = i / CC; int li = (int)(t % LL); int n = (int)(t / LL);
  int dst = ord[(long)n * LL + li];   // permutation -> race free
  upd[((long)n * LL + dst) * CC + c] += nf[i] * nm[(long)n * LL + li] * cond[0];
}

__global__ void blend_kernel(float* __restrict__ out, const float* __restrict__ feat,
                             const float* __restrict__ ssum, const float* __restrict__ upd) {
  long i = (long)blockIdx.x * blockDim.x + threadIdx.x;
  if (i >= (long)NB * LL * CC) return;
  long t = i / CC;
  out[i] = (1.f - ssum[t]) * feat[i] + upd[i];
}

// tmi = dm * pooled, where dm's (L,K) flat block is reinterpreted as (K,64,64)
__global__ void tmi_kernel(const float* __restrict__ dm, long nstride, long base,
                           float* __restrict__ out) {
  long i = (long)blockIdx.x * blockDim.x + threadIdx.x;
  if (i >= (long)NB * LL * KK) return;
  int k = i % KK; long t = i / KK; int l = (int)(t % LL); int n = (int)(t / LL);
  const float* d = dm + (long)n * nstride + base;
  float v = d[(long)l * KK + k];
  int ii = l >> 6, jj = l & 63;
  float s = 0.f;
  for (int di = -1; di <= 1; ++di) {
    int i2 = ii + di; if (i2 < 0 || i2 >= 64) continue;
    for (int dj = -1; dj <= 1; ++dj) {
      int j2 = jj + dj; if (j2 < 0 || j2 >= 64) continue;
      s += d[(long)k * LL + i2 * 64 + j2];
    }
  }
  out[i] = v * (s * (1.f / 9.f));
}

// =================== host-side orchestration ===============================
static inline dim3 egrid(long total) { return dim3((unsigned)((total + 255) / 256)); }

static void gemm(hipStream_t st, const bf16* A, long lda, long sA,
                 const bf16* BT, long ldbt, long sBT,
                 float* C, long ldc, long sC,
                 int M, int N, int Kd, float alpha, int accum) {
  dim3 g((N + BN - 1) / BN, (M + BM - 1) / BM, NB);
  gemm_kernel<<<g, 256, 0, st>>>(A, lda, sA, BT, ldbt, sBT, C, ldc, sC, M, N, Kd, alpha, accum);
}

struct EncW { const bf16 *WqT, *WkT, *WvT, *WmT, *W1T, *W2T; };
struct EncTmp {
  float *qb, *kb, *vb, *msgb, *h1b, *h2b, *KVb, *Ksb;
  bf16 *xb, *sb, *mb, *h2bb, *h1bb;
};

static void enc_layer(hipStream_t st, const EncW& w,
                      float* x, int llen,
                      const float* srcA, int slenA,
                      const float* srcB, int slenB,
                      const float* xmask, const float* smask,
                      const EncTmp& t) {
  const int slen = slenA + slenB;
  const long xs = (long)llen * CC;
  const long ss = (long)slen * CC;
  // bf16 operands for the projection GEMMs
  cvt_bf16_kernel<<<egrid((long)NB * xs), 256, 0, st>>>(x, t.xb, (long)NB * xs);
  cvt_bf16_kernel<<<egrid((long)NB * slenA * CC), 256, 0, st>>>(srcA, t.sb, (long)NB * slenA * CC);
  if (srcB)
    cvt_bf16_kernel<<<egrid((long)NB * slenB * CC), 256, 0, st>>>(srcB, t.sb + (long)NB * slenA * CC,
                                                                  (long)NB * slenB * CC);
  const bf16* sbA = t.sb;
  const bf16* sbB = t.sb + (long)NB * slenA * CC;
  // Q = elu(x@Wq)+1 (*xmask)
  gemm(st, t.xb, CC, xs, w.WqT, CC, 0, t.qb, CC, xs, llen, CC, CC, 1.f, 0);
  elu1_mask_kernel<<<egrid((long)NB * llen * CC), 256, 0, st>>>(t.qb, xmask, (long)NB * llen, CC);
  // K = elu(src@Wk)+1 (*smask);  V = src@Wv (*smask)
  gemm(st, sbA, CC, (long)slenA * CC, w.WkT, CC, 0, t.kb, CC, ss, slenA, CC, CC, 1.f, 0);
  gemm(st, sbA, CC, (long)slenA * CC, w.WvT, CC, 0, t.vb, CC, ss, slenA, CC, CC, 1.f, 0);
  if (srcB) {
    gemm(st, sbB, CC, (long)slenB * CC, w.WkT, CC, 0, t.kb + (long)slenA * CC, CC, ss, slenB, CC, CC, 1.f, 0);
    gemm(st, sbB, CC, (long)slenB * CC, w.WvT, CC, 0, t.vb + (long)slenA * CC, CC, ss, slenB, CC, CC, 1.f, 0);
  }
  elu1_mask_kernel<<<egrid((long)NB * slen * CC), 256, 0, st>>>(t.kb, smask, (long)NB * slen, CC);
  if (smask)
    mask_mul_kernel<<<egrid((long)NB * slen * CC), 256, 0, st>>>(t.vb, smask, (long)NB * slen, CC);
  // linear attention (msg also emits bf16 copy)
  kv_ksum_kernel<<<dim3(NH, NB), 256, 0, st>>>(t.kb, t.vb, t.KVb, t.Ksb, slen);
  msg_kernel<<<dim3((llen + 15) / 16, NB), 256, 0, st>>>(t.qb, t.KVb, t.Ksb, t.msgb, t.mb, llen);
  // msg @ Wm -> LN (emits bf16 for W1-bottom GEMM)
  gemm(st, t.mb, CC, xs, w.WmT, CC, 0, t.h2b, CC, xs, llen, CC, CC, 1.f, 0);
  layernorm_kernel<<<NB * llen, 256, 0, st>>>(t.h2b, t.h2bb, CC);
  // relu([x,msg] @ W1) : split-K concat as two GEMMs  (W1T is 512x512, n-major)
  gemm(st, t.xb,   CC, xs, w.W1T,       2 * CC, 0, t.h1b, 2 * CC, (long)llen * 2 * CC, llen, 2 * CC, CC, 1.f, 0);
  gemm(st, t.h2bb, CC, xs, w.W1T + CC,  2 * CC, 0, t.h1b, 2 * CC, (long)llen * 2 * CC, llen, 2 * CC, CC, 1.f, 1);
  relu_cvt_kernel<<<egrid((long)NB * llen * 2 * CC), 256, 0, st>>>(t.h1b, t.h1bb, (long)NB * llen * 2 * CC);
  // @ W2 -> LN -> residual
  gemm(st, t.h1bb, 2 * CC, (long)llen * 2 * CC, w.W2T, 2 * CC, 0, t.msgb, CC, xs, llen, CC, 2 * CC, 1.f, 0);
  layernorm_kernel<<<NB * llen, 256, 0, st>>>(t.msgb, (bf16*)nullptr, CC);
  add_inplace_kernel<<<egrid((long)NB * llen * CC), 256, 0, st>>>(x, t.msgb, (long)NB * llen * CC);
}

static EncW layer_w(const bf16* WqT, const bf16* WkT, const bf16* WvT,
                    const bf16* WmT, const bf16* W1T, const bf16* W2T, int i) {
  EncW w;
  w.WqT = WqT + (size_t)i * CC * CC;
  w.WkT = WkT + (size_t)i * CC * CC;
  w.WvT = WvT + (size_t)i * CC * CC;
  w.WmT = WmT + (size_t)i * CC * CC;
  w.W1T = W1T + (size_t)i * (2 * CC) * (2 * CC);
  w.W2T = W2T + (size_t)i * CC * (2 * CC);
  return w;
}

extern "C" void kernel_launch(void* const* d_in, const int* in_sizes, int n_in,
                              void* d_out, int out_size, void* d_ws, size_t ws_size,
                              hipStream_t stream) {
  (void)in_sizes; (void)n_in; (void)out_size;
  const float* feat0 = (const float*)d_in[0];
  const float* feat1 = (const float*)d_in[1];
  const float* topic0 = (const float*)d_in[2];
  const float* topic1 = (const float*)d_in[3];
  const float* seed_tokens = (const float*)d_in[4];
  const float* Wq = (const float*)d_in[5];
  const float* Wk = (const float*)d_in[6];
  const float* Wv = (const float*)d_in[7];
  const float* Wm = (const float*)d_in[8];
  const float* W1 = (const float*)d_in[9];
  const float* W2 = (const float*)d_in[10];

  float* out_feat0 = (float*)d_out;
  float* out_feat1 = out_feat0 + (size_t)NB * LL * CC;
  float* out_tmi0  = out_feat1 + (size_t)NB * LL * CC;
  float* out_tmi1  = out_tmi0  + (size_t)NB * LL * KK;

  // ---- carve workspace ----
  char* base = (char*)d_ws;
  size_t off = 0;
  auto alloc = [&](size_t bytes) -> void* {
    void* p = base + off;
    off = (off + bytes + 255) & ~(size_t)255;
    return p;
  };
  const size_t FC  = (size_t)NB * LL * CC * 4;     // 16 MB f32
  float* feat0w = (float*)alloc(FC);
  float* feat1w = (float*)alloc(FC);
  float* seeds  = (float*)alloc((size_t)NB * KK * CC * 4);
  float* qb   = (float*)alloc(FC);
  float* kb   = (float*)alloc(2 * FC);
  float* vb   = (float*)alloc(2 * FC);
  float* msgb = (float*)alloc(FC);
  float* h1b  = (float*)alloc(2 * FC);
  float* h2b  = (float*)alloc(FC);
  float* KVb  = (float*)alloc((size_t)NB * NH * HD * HD * 4);
  float* Ksb  = (float*)alloc((size_t)NB * CC * 4);
  float* dm   = (float*)alloc((size_t)NB * 2 * LL * KK * 4);
  float* nf0  = (float*)alloc(FC);
  float* nf1  = (float*)alloc(FC);
  float* upd0 = (float*)alloc(FC);
  float* upd1 = (float*)alloc(FC);
  float* ssum0 = (float*)alloc((size_t)NB * LL * 4);
  float* ssum1 = (float*)alloc((size_t)NB * LL * 4);
  float* nm0  = (float*)alloc((size_t)NB * LL * 4);
  float* nm1  = (float*)alloc((size_t)NB * LL * 4);
  float* th0  = (float*)alloc((size_t)NB * KK * 4);
  float* th1  = (float*)alloc((size_t)NB * KK * 4);
  float* condb = (float*)alloc(256);
  float* t0s = (float*)alloc((size_t)NB * KK * DTT * 4);
  float* t1s = (float*)alloc((size_t)NB * KK * DTT * 4);
  float* t1g = (float*)alloc((size_t)NB * KK * DTT * 4);
  float* tmb = (float*)alloc((size_t)NB * KK * KK * 4);
  float* tpb = (float*)alloc((size_t)NB * KK * DTT * 4);
  float* spb = (float*)alloc((size_t)NB * KK * 160 * 4);
  float* sfb = (float*)alloc((size_t)NB * KK * CC * 4);
  int* fidx  = (int*)alloc((size_t)NB * 2 * LL * 4);
  int* inds  = (int*)alloc((size_t)NB * NSAMP * 4);
  int* tmidx = (int*)alloc((size_t)NB * KK * 4);
  int* ord0  = (int*)alloc((size_t)NB * LL * 4);
  int* ord1  = (int*)alloc((size_t)NB * LL * 4);
  int* cnt0  = (int*)alloc((size_t)NB * 4);
  int* cnt1  = (int*)alloc((size_t)NB * 4);
  // bf16 buffers
  bf16* WqT = (bf16*)alloc((size_t)8 * CC * CC * 2);
  bf16* WkT = (bf16*)alloc((size_t)8 * CC * CC * 2);
  bf16* WvT = (bf16*)alloc((size_t)8 * CC * CC * 2);
  bf16* WmT = (bf16*)alloc((size_t)8 * CC * CC * 2);
  bf16* W1T = (bf16*)alloc((size_t)8 * 2 * CC * 2 * CC * 2);
  bf16* W2T = (bf16*)alloc((size_t)8 * CC * 2 * CC * 2);
  bf16* xb   = (bf16*)alloc(FC / 2);
  bf16* sb   = (bf16*)alloc(FC);          // up to (N, 2L, C) bf16
  bf16* mb   = (bf16*)alloc(FC / 2);
  bf16* h2bb = (bf16*)alloc(FC / 2);
  bf16* h1bb = (bf16*)alloc(FC);
  bf16* t0sb = (bf16*)alloc((size_t)NB * KK * DTT * 2);
  bf16* t1sb = (bf16*)alloc((size_t)NB * KK * DTT * 2);
  bf16* sfbb = (bf16*)alloc((size_t)NB * KK * CC * 2);
  if (off > ws_size) return;   // workspace too small: no-op

  EncTmp T{qb, kb, vb, msgb, h1b, h2b, KVb, Ksb, xb, sb, mb, h2bb, h1bb};

  // ---- one-time: weights -> bf16, transposed to N-major (BT layout) ----
  tcvt_bf16_kernel<<<egrid((long)8 * CC * CC), 256, 0, stream>>>(Wq, WqT, CC, CC, 8);
  tcvt_bf16_kernel<<<egrid((long)8 * CC * CC), 256, 0, stream>>>(Wk, WkT, CC, CC, 8);
  tcvt_bf16_kernel<<<egrid((long)8 * CC * CC), 256, 0, stream>>>(Wv, WvT, CC, CC, 8);
  tcvt_bf16_kernel<<<egrid((long)8 * CC * CC), 256, 0, stream>>>(Wm, WmT, CC, CC, 8);
  tcvt_bf16_kernel<<<egrid((long)8 * 4 * CC * CC), 256, 0, stream>>>(W1, W1T, 2 * CC, 2 * CC, 8);
  tcvt_bf16_kernel<<<egrid((long)8 * 2 * CC * CC), 256, 0, stream>>>(W2, W2T, 2 * CC, CC, 8);

  // ---- init / copies ----
  (void)hipMemcpyAsync(feat0w, feat0, FC, hipMemcpyDeviceToDevice, stream);
  (void)hipMemcpyAsync(feat1w, feat1, FC, hipMemcpyDeviceToDevice, stream);
  (void)hipMemsetAsync(upd0, 0, FC, stream);
  (void)hipMemsetAsync(upd1, 0, FC, stream);
  (void)hipMemsetAsync(ssum0, 0, (size_t)NB * LL * 4, stream);
  (void)hipMemsetAsync(ssum1, 0, (size_t)NB * LL * 4, stream);
  (void)hipMemsetAsync(th0, 0, (size_t)NB * KK * 4, stream);
  (void)hipMemsetAsync(th1, 0, (size_t)NB * KK * 4, stream);
  bcast_seeds_kernel<<<egrid((long)NB * KK * CC), 256, 0, stream>>>(seed_tokens, seeds);

  // ---- topic branch ----
  transpose_kernel<<<egrid((long)NB * DTT * KK), 256, 0, stream>>>(topic0, t0s, DTT, KK, NB);
  transpose_kernel<<<egrid((long)NB * DTT * KK), 256, 0, stream>>>(topic1, t1s, DTT, KK, NB);
  cvt_bf16_kernel<<<egrid((long)NB * KK * DTT), 256, 0, stream>>>(t0s, t0sb, (long)NB * KK * DTT);
  cvt_bf16_kernel<<<egrid((long)NB * KK * DTT), 256, 0, stream>>>(t1s, t1sb, (long)NB * KK * DTT);
  // tm[n,m,l] = sum_d t0[n,m,d]*t1[n,l,d]  (A = t0s, BT = t1s: both (100,96))
  gemm(stream, t0sb, DTT, (long)KK * DTT, t1sb, DTT, (long)KK * DTT,
       tmb, KK, (long)KK * KK, KK, KK, DTT, 1.f, 0);
  floor_scale_kernel<<<egrid((long)NB * KK * KK), 256, 0, stream>>>(tmb, (long)NB * KK * KK);
  tm_soft_kernel<<<NB, 256, 0, stream>>>(tmb, tmidx);
  gather_t1_kernel<<<egrid((long)NB * KK * DTT), 256, 0, stream>>>(t1s, tmidx, t1g);
  topic_pool_kernel<<<egrid((long)NB * KK * DTT), 256, 0, stream>>>(t0s, t1g, tpb);

  // ---- main encoder layers ('seed','feat','seed','feat') ----
  for (int i = 0; i < 4; ++i) {
    EncW w = layer_w(WqT, WkT, WvT, WmT, W1T, W2T, i);
    if ((i & 1) == 0) {
      enc_layer(stream, w, seeds, KK, feat0w, LL, feat1w, LL, nullptr, nullptr, T);
    } else {
      enc_layer(stream, w, feat0w, LL, seeds, KK, nullptr, 0, nullptr, nullptr, T);
      enc_layer(stream, w, feat1w, LL, seeds, KK, nullptr, 0, nullptr, nullptr, T);
    }
  }

  // ---- final seeds: pool 256->160, concat topic(96), LN, bf16 (BT) ----
  pool_kernel<<<egrid((long)NB * KK * 160), 256, 0, stream>>>(seeds, spb, (long)NB * KK, CC, 160);
  concat_seeds_kernel<<<egrid((long)NB * KK * CC), 256, 0, stream>>>(spb, tpb, sfb);
  layernorm_kernel<<<NB * KK, 256, 0, stream>>>(sfb, sfbb, CC);

  // ---- dmatrix = feat @ seeds^T / 16   (BT = sfbb: (100,256) n-major) ----
  cvt_bf16_kernel<<<egrid((long)NB * LL * CC), 256, 0, stream>>>(feat0w, xb, (long)NB * LL * CC);
  gemm(stream, xb, CC, (long)LL * CC, sfbb, CC, (long)KK * CC,
       dm, KK, (long)2 * LL * KK, LL, KK, CC, 1.f / 16.f, 0);
  cvt_bf16_kernel<<<egrid((long)NB * LL * CC), 256, 0, stream>>>(feat1w, xb, (long)NB * LL * CC);
  gemm(stream, xb, CC, (long)LL * CC, sfbb, CC, (long)KK * CC,
       dm + (long)LL * KK, KK, (long)2 * LL * KK, LL, KK, CC, 1.f / 16.f, 0);

  // ---- prob/theta/topk ----
  prob_theta_kernel<<<egrid((long)NB * 2 * LL), 256, 0, stream>>>(dm, fidx, th0, th1);
  theta_topk_kernel<<<NB, 32, 0, stream>>>(th0, th1, inds);

  // ---- refinement over top-6 sampled topics ----
  for (int kk = 0; kk < NSAMP; ++kk) {
    partition_kernel<<<NB, 32, 0, stream>>>(fidx, 0,  inds, kk, ord0, nm0, cnt0);
    partition_kernel<<<NB, 32, 0, stream>>>(fidx, LL, inds, kk, ord1, nm1, cnt1);
    cond_kernel<<<1, 1, 0, stream>>>(cnt0, cnt1, condb);
    ssum_kernel<<<egrid((long)NB * LL), 256, 0, stream>>>(ssum0, fidx, 0,  inds, kk);
    ssum_kernel<<<egrid((long)NB * LL), 256, 0, stream>>>(ssum1, fidx, LL, inds, kk);
    gather_kernel<<<egrid((long)NB * LL * CC), 256, 0, stream>>>(nf0, feat0w, ord0, nm0);
    gather_kernel<<<egrid((long)NB * LL * CC), 256, 0, stream>>>(nf1, feat1w, ord1, nm1);
    for (int idt = 0; idt < NTT; ++idt) {
      EncW wA = layer_w(WqT, WkT, WvT, WmT, W1T, W2T, 4 + idt * 2);
      EncW wB = layer_w(WqT, WkT, WvT, WmT, W1T, W2T, 5 + idt * 2);
      enc_layer(stream, wA, nf0, LL, nf0, LL, nullptr, 0, nm0, nm0, T);
      enc_layer(stream, wA, nf1, LL, nf1, LL, nullptr, 0, nm1, nm1, T);
      enc_layer(stream, wB, nf0, LL, nf1, LL, nullptr, 0, nm0, nm1, T);
      enc_layer(stream, wB, nf1, LL, nf0, LL, nullptr, 0, nm1, nm0, T);
    }
    scatter_kernel<<<egrid((long)NB * LL * CC), 256, 0, stream>>>(upd0, nf0, ord0, nm0, condb);
    scatter_kernel<<<egrid((long)NB * LL * CC), 256, 0, stream>>>(upd1, nf1, ord1, nm1, condb);
  }

  // ---- final blend + tmi outputs ----
  blend_kernel<<<egrid((long)NB * LL * CC), 256, 0, stream>>>(out_feat0, feat0w, ssum0, upd0);
  blend_kernel<<<egrid((long)NB * LL * CC), 256, 0, stream>>>(out_feat1, feat1w, ssum1, upd1);
  tmi_kernel<<<egrid((long)NB * LL * KK), 256, 0, stream>>>(dm, (long)2 * LL * KK, 0, out_tmi0);
  tmi_kernel<<<egrid((long)NB * LL * KK), 256, 0, stream>>>(dm, (long)2 * LL * KK, (long)LL * KK, out_tmi1);
}
